// DeiTSelfAttentionMask_67310727463108
// MI455X (gfx1250) — compile-verified, matrix operand order checked by reading, not run on hardware
//
#include <hip/hip_runtime.h>

#define B_   16
#define S_   577
#define HID_ 768
#define H_   12
#define D_   64
#define NK_  576
#define BS_  (B_*S_)      /* 9232 rows of the QKV GEMM */
#define SPAD 640          /* padded sequence for head-layout buffers */

typedef __bf16 bf16;
typedef __attribute__((ext_vector_type(16))) __bf16 v16bf;
typedef __attribute__((ext_vector_type(8)))  __bf16 v8bf;
typedef __attribute__((ext_vector_type(8)))  float  v8f;

// Load a 16-element A/B WMMA fragment slice for one lane:
// elements p[0..7] and p[16..23]  (K-slots base..base+7 and base+16..base+23)
__device__ __forceinline__ v16bf ld_frag(const bf16* p) {
  v8bf lo = *(const v8bf*)p;
  v8bf hi = *(const v8bf*)(p + 16);
  v16bf r;
#pragma unroll
  for (int i = 0; i < 8; ++i) { r[i] = lo[i]; r[i + 8] = hi[i]; }
  return r;
}

__device__ __forceinline__ v8f wmma_bf16(v16bf a, v16bf b, v8f c) {
  return __builtin_amdgcn_wmma_f32_16x16x32_bf16(false, a, false, b, (short)0, c,
                                                 false, false);
}

__device__ __forceinline__ float rmax16(float x) {
#pragma unroll
  for (int m = 1; m < 16; m <<= 1) x = fmaxf(x, __shfl_xor(x, m, 16));
  return x;
}

// ---------------------------------------------------------------- inv map ---
__global__ void build_inv(const int* __restrict__ vidx, int* __restrict__ inv) {
  int t = threadIdx.x;                 // 768 threads
  inv[t] = -1;
  __syncthreads();
  if (t < NK_) inv[vidx[t]] = t;
}

// ------------------------------------------------------------- QKV GEMM -----
// C[9232,576] = hs[9232,768] @ W + b, scattered via idx into bf16 head layout
// dst[b, h, s(SPAD), d].  Block: 64x64 tile, 8 waves each 16x32, K-step 32.
__global__ __launch_bounds__(256) void qkv_gemm(
    const float* __restrict__ hs,
    const float* __restrict__ Wq, const float* __restrict__ bq,
    const int* __restrict__ qidx, bf16* __restrict__ Qh,
    const float* __restrict__ Wk, const float* __restrict__ bk,
    const int* __restrict__ kidx, bf16* __restrict__ Kh,
    const float* __restrict__ Wv, const float* __restrict__ bv,
    const int* __restrict__ vidx, bf16* __restrict__ Vh) {
  const float* W; const float* bias; const int* idx; bf16* dst;
  if (blockIdx.z == 0)      { W = Wq; bias = bq; idx = qidx; dst = Qh; }
  else if (blockIdx.z == 1) { W = Wk; bias = bk; idx = kidx; dst = Kh; }
  else                      { W = Wv; bias = bv; idx = vidx; dst = Vh; }

  __shared__ bf16 sA[64][32];    // [m][k]
  __shared__ bf16 sBt[64][32];   // [n][k] (W tile transposed)

  const int tid  = threadIdx.x;
  const int w    = tid >> 5, l = tid & 31;
  const int hb   = l >> 4, ln = l & 15, base = hb * 8;
  const int mbase = blockIdx.x * 64;
  const int nbase = blockIdx.y * 64;
  const int subm = (w >> 1) * 16, subn = (w & 1) * 32;

  // staging coordinates
  const int ar = tid >> 2, acs = (tid & 3) * 8;   // A: row 0..63, k-slice
  const int agm = mbase + ar;
  const bool avalid = (agm < BS_);
  const int bn = tid & 63, bks = (tid >> 6) * 8;  // B: n 0..63, k-slice

  v8f acc[2];
#pragma unroll
  for (int c = 0; c < 2; ++c)
#pragma unroll
    for (int r = 0; r < 8; ++r) acc[c][r] = 0.0f;

  for (int k0 = 0; k0 < HID_; k0 += 32) {
    __syncthreads();
    {   // stage A: 64x32 fp32 -> bf16, one b128 LDS store per thread
      v8bf pk;
      if (avalid) {
        const float* src = hs + (size_t)agm * HID_ + k0 + acs;
        float4 f0 = *(const float4*)src;
        float4 f1 = *(const float4*)(src + 4);
        pk[0] = (bf16)f0.x; pk[1] = (bf16)f0.y;
        pk[2] = (bf16)f0.z; pk[3] = (bf16)f0.w;
        pk[4] = (bf16)f1.x; pk[5] = (bf16)f1.y;
        pk[6] = (bf16)f1.z; pk[7] = (bf16)f1.w;
        if (k0 + 32 < HID_) __builtin_prefetch(src + 32, 0, 1);
      } else {
#pragma unroll
        for (int i = 0; i < 8; ++i) pk[i] = (bf16)0.0f;
      }
      *(v8bf*)&sA[ar][acs] = pk;
    }
    {   // stage B transposed: thread owns row n of sBt, 8 consecutive k
      const float* src = W + (size_t)(k0 + bks) * NK_ + nbase + bn;
      v8bf pk;
#pragma unroll
      for (int i = 0; i < 8; ++i) pk[i] = (bf16)src[(size_t)i * NK_];
      if (k0 + 32 < HID_) __builtin_prefetch(src + (size_t)32 * NK_, 0, 1);
      *(v8bf*)&sBt[bn][bks] = pk;
    }
    __syncthreads();

    v16bf a = ld_frag(&sA[subm + ln][base]);
#pragma unroll
    for (int c = 0; c < 2; ++c) {
      v16bf b = ld_frag(&sBt[subn + c * 16 + ln][base]);
      acc[c] = wmma_bf16(a, b, acc[c]);
    }
  }

  // epilogue: +bias, scatter kept column n -> channel idx[n] -> (head, d)
#pragma unroll
  for (int c = 0; c < 2; ++c) {
    int ng = nbase + subn + c * 16 + ln;
    float bv_ = bias[ng];
    int   ci  = idx[ng];
    int   hh  = ci >> 6, dd = ci & 63;
#pragma unroll
    for (int r = 0; r < 8; ++r) {
      int mloc = subm + r + 8 * hb;
      int m    = mbase + mloc;
      if (m < BS_) {
        unsigned bb = (unsigned)m / S_;
        unsigned ss = (unsigned)m - bb * S_;
        dst[(((size_t)bb * H_ + hh) * SPAD + ss) * D_ + dd] =
            (bf16)(acc[c][r] + bv_);
      }
    }
  }
}

// ------------------------------------------------------------- attention ----
// Block = (q-tile of 128, b*H+h).  8 waves, 16 queries each.
// Flash-attention over key-tile pairs (32 keys -> K=32 WMMA for P.V).
// - P.V uses permuted K-slots sigma(p)=(p>>1)+(p&1)*16 so each lane's (p0,p1)
//   packs into one b32 LDS store; V staged with the same permutation.
// - Softmax denominator computed on the matrix pipe: a 5th accumulator chunk
//   multiplies P by a ones-column, tracking the online row-sum exactly.
// - Key masking only in the final pair (keys >= 577).
__global__ __launch_bounds__(256) void attn(
    const bf16* __restrict__ Qh, const bf16* __restrict__ Kh,
    const bf16* __restrict__ Vh, const int* __restrict__ inv_v,
    float* __restrict__ out) {
  __shared__ bf16 sK[32][64];        // K tile, row-major [key][d]
  __shared__ bf16 sVt[64][32];       // V tile, [d][slot] with slot permutation
  __shared__ bf16 sP[8][16][32];     // per-wave P bounce, [m][slot]

  const int tid = threadIdx.x;
  const int w = tid >> 5, l = tid & 31;
  const int hb = l >> 4, ln = l & 15, base = hb * 8;
  const int bh = blockIdx.y;
  const int b  = bh / H_, h = bh % H_;
  const int q0 = blockIdx.x * 128;

  const bf16* Qb = Qh + (size_t)bh * SPAD * D_;
  const bf16* Kb = Kh + (size_t)bh * SPAD * D_;
  const bf16* Vb = Vh + (size_t)bh * SPAD * D_;

  const int qm = q0 + w * 16 + ln;            // < SPAD by construction
  v16bf aQ0 = ld_frag(Qb + (size_t)qm * D_ + base);
  v16bf aQ1 = ld_frag(Qb + (size_t)qm * D_ + 32 + base);

  v16bf ones;
#pragma unroll
  for (int i = 0; i < 16; ++i) ones[i] = (bf16)1.0f;

  // V staging: thread owns row d of sVt, 8 consecutive slots
  const int vd = tid & 63, vks = (tid >> 6) * 8;

  v8f acc[5];                         // [0..3]: d-chunks, [4]: row-sum (lrun)
  float mrun[8];
#pragma unroll
  for (int c = 0; c < 5; ++c)
#pragma unroll
    for (int r = 0; r < 8; ++r) acc[c][r] = 0.0f;
#pragma unroll
  for (int r = 0; r < 8; ++r) mrun[r] = -1e30f;

  // base-2 softmax: fold 1/sqrt(64) and log2(e) into one scale
  const float scale2 = 0.125f * 1.44269504089f;

  auto pair_step = [&](const int kb0, const bool maskKeys)
      __attribute__((always_inline)) {
    __syncthreads();
    {   // stage K tile: flat b128 copy (32x64 bf16 = 2048 elems / 256 thr)
      ((v8bf*)sK)[tid] = *(const v8bf*)(Kb + (size_t)kb0 * D_ + tid * 8);
    }
    {   // stage V tile transposed with slot permutation; one b128 store
      v8bf pk;
#pragma unroll
      for (int i = 0; i < 8; ++i) {
        int pos = vks + i;
        int key = kb0 + (pos >> 1) + (pos & 1) * 16;   // sigma(pos)
        pk[i] = Vb[(size_t)key * D_ + vd];
      }
      *(v8bf*)&sVt[vd][vks] = pk;
    }
    // prefetch next tiles (speculative; padded buffers keep it in-bounds)
    __builtin_prefetch(Kb + (size_t)(kb0 + 32) * D_ + tid * 8, 0, 1);
    __builtin_prefetch(Vb + (size_t)(kb0 + 32 + vks) * D_ + vd, 0, 1);
    __syncthreads();

    // scores for the two 16-key tiles of this pair
    v8f s[2];
#pragma unroll
    for (int t = 0; t < 2; ++t) {
      v16bf bK0 = ld_frag(&sK[t * 16 + ln][base]);
      v16bf bK1 = ld_frag(&sK[t * 16 + ln][32 + base]);
      v8f z;
#pragma unroll
      for (int r = 0; r < 8; ++r) z[r] = 0.0f;
      z = wmma_bf16(aQ0, bK0, z);
      z = wmma_bf16(aQ1, bK1, z);
      if (maskKeys) {
        bool valid = (kb0 + t * 16 + ln) < S_;
#pragma unroll
        for (int r = 0; r < 8; ++r) s[t][r] = valid ? z[r] * scale2 : -1e30f;
      } else {
#pragma unroll
        for (int r = 0; r < 8; ++r) s[t][r] = z[r] * scale2;
      }
    }

    // online softmax in base 2 (row = r + 8*hb, cols across 16-lane half)
#pragma unroll
    for (int r = 0; r < 8; ++r) {
      float tm   = rmax16(fmaxf(s[0][r], s[1][r]));
      float mnew = fmaxf(mrun[r], tm);
      float al   = __builtin_amdgcn_exp2f(mrun[r] - mnew);
      float p0   = __builtin_amdgcn_exp2f(s[0][r] - mnew);
      float p1   = __builtin_amdgcn_exp2f(s[1][r] - mnew);
      mrun[r] = mnew;
#pragma unroll
      for (int c = 0; c < 5; ++c) acc[c][r] *= al;
      // keys ln and ln+16 -> adjacent permuted slots 2*ln, 2*ln+1
      union { bf16 hh[2]; unsigned u; } pk;
      pk.hh[0] = (bf16)p0; pk.hh[1] = (bf16)p1;
      *(unsigned*)&sP[w][r + 8 * hb][2 * ln] = pk.u;
    }
    // same-wave LDS write->read ordering (transpose bounce)
    asm volatile("s_wait_dscnt 0" ::: "memory");

    v16bf aP = ld_frag(&sP[w][ln][base]);
#pragma unroll
    for (int c = 0; c < 4; ++c) {
      v16bf bV = ld_frag(&sVt[c * 16 + ln][base]);
      acc[c] = wmma_bf16(aP, bV, acc[c]);
    }
    acc[4] = wmma_bf16(aP, ones, acc[4]);   // online row-sum on matrix pipe
  };

  for (int pair = 0; pair < 18; ++pair) pair_step(pair * 32, false);
  pair_step(18 * 32, true);               // keys 576..607: mask >= 577

  // normalize + gather kept channels of ctx into out[B,S,576]
#pragma unroll
  for (int c = 0; c < 4; ++c) {
    int d  = c * 16 + ln;
    int j  = inv_v[h * 64 + d];
#pragma unroll
    for (int r = 0; r < 8; ++r) {
      int mloc = r + 8 * hb;
      int q    = q0 + w * 16 + mloc;
      if (q < S_ && j >= 0)
        out[((size_t)b * S_ + q) * NK_ + j] =
            acc[c][r] * __builtin_amdgcn_rcpf(acc[4][r]);
    }
  }
}

// ---------------------------------------------------------------- launch ----
extern "C" void kernel_launch(void* const* d_in, const int* in_sizes, int n_in,
                              void* d_out, int out_size, void* d_ws,
                              size_t ws_size, hipStream_t stream) {
  const float* hs  = (const float*)d_in[0];
  const float* Wq  = (const float*)d_in[1];
  const float* bq  = (const float*)d_in[2];
  const float* Wk  = (const float*)d_in[3];
  const float* bk  = (const float*)d_in[4];
  const float* Wv  = (const float*)d_in[5];
  const float* bv  = (const float*)d_in[6];
  const int*   qix = (const int*)d_in[7];
  const int*   kix = (const int*)d_in[8];
  const int*   vix = (const int*)d_in[9];
  float* out = (float*)d_out;

  const size_t headElems = (size_t)B_ * H_ * SPAD * D_;   // 7,864,320
  char* ws = (char*)d_ws;
  bf16* Qh = (bf16*)ws;
  bf16* Kh = Qh + headElems;
  bf16* Vh = Kh + headElems;
  int*  inv = (int*)(ws + 3 * headElems * sizeof(bf16));

  // zero the padded head-layout buffers (implements pruning scatter + padding)
  hipMemsetAsync(ws, 0, 3 * headElems * sizeof(bf16), stream);

  build_inv<<<1, HID_, 0, stream>>>(vix, inv);

  dim3 gGemm((BS_ + 63) / 64, NK_ / 64, 3);               // 145 x 9 x 3
  qkv_gemm<<<gGemm, 256, 0, stream>>>(hs, Wq, bq, qix, Qh, Wk, bk, kix, Kh, Wv,
                                      bv, vix, Vh);

  dim3 gAttn((S_ + 127) / 128, B_ * H_);                  // 5 x 192
  attn<<<gAttn, 256, 0, stream>>>(Qh, Kh, Vh, inv, out);
}
